// FeedbackTransformerModel_32727650795906
// MI455X (gfx1250) — matrix operand order repaired and checked by reading, not compiled, not run on hardware
//
#include <hip/hip_runtime.h>
#include <hip/hip_bf16.h>

#define D      1024
#define FF     4096
#define NH     16
#define HD     64
#define NL     4
#define BB     4
#define SS     512
#define VO     32000
#define NWG    64        // persistent workgroups for the scan kernel

typedef __attribute__((ext_vector_type(16))) __bf16 v16bf;
typedef __attribute__((ext_vector_type(8)))  __bf16 v8bf;
typedef __attribute__((ext_vector_type(8)))  float  v8f;
typedef __attribute__((ext_vector_type(4)))  unsigned int v4u;
typedef __attribute__((ext_vector_type(8)))  int v8i;
typedef __attribute__((ext_vector_type(4)))  int v4i;

#ifndef __has_builtin
#define __has_builtin(x) 0
#endif
#if __has_builtin(__builtin_amdgcn_tensor_load_to_lds) && \
    __has_builtin(__builtin_amdgcn_s_wait_tensorcnt)
#define HAVE_TDM 1
#else
#define HAVE_TDM 0
#endif

// ---------------------------------------------------------------- helpers
__device__ __forceinline__ __bf16 tobf(float x) {
    __hip_bfloat16 h = __float2bfloat16(x);
    return *reinterpret_cast<__bf16*>(&h);
}

__device__ __forceinline__ float block_sum(float v, float* s_red) {
    int tid = threadIdx.x;
    s_red[tid] = v; __syncthreads();
    for (int off = 128; off > 0; off >>= 1) {
        if (tid < off) s_red[tid] += s_red[tid + off];
        __syncthreads();
    }
    float r = s_red[0]; __syncthreads();
    return r;
}

__device__ __forceinline__ float block_max(float v, float* s_red) {
    int tid = threadIdx.x;
    s_red[tid] = v; __syncthreads();
    for (int off = 128; off > 0; off >>= 1) {
        if (tid < off) s_red[tid] = fmaxf(s_red[tid], s_red[tid + off]);
        __syncthreads();
    }
    float r = s_red[0]; __syncthreads();
    return r;
}

// Hierarchical device-wide barrier: 8 groups x 8 WGs on separate lines,
// group winners meet on a root line, generation broadcast for release.
__device__ __forceinline__ void gbar(int* bar) {
    __syncthreads();
    if (threadIdx.x == 0) {
        __threadfence();
        volatile int* vgen = bar + 288;
        int g = *vgen;
        int grp = (int)blockIdx.x >> 3;
        int p = atomicAdd(bar + grp * 32, 1);
        if (p == 7) {
            bar[grp * 32] = 0;
            int q = atomicAdd(bar + 256, 1);
            if (q == 7) {
                bar[256] = 0;
                __threadfence();
                atomicAdd(bar + 288, 1);
            } else {
                while (*vgen == g) __builtin_amdgcn_s_sleep(1);
            }
        } else {
            while (*vgen == g) __builtin_amdgcn_s_sleep(1);
        }
        __threadfence();
    }
    __syncthreads();
}

// 16x16 WMMA tile with software-pipelined K-loop. A rows come from LDS
// (bf16, row length K, rows >= BB read a zero block); B from bf16 W^T (NxK).
__device__ __forceinline__ v8f wmma_tile16(const __bf16* sA, const __bf16* s_zero,
                                           const __hip_bfloat16* WT, int K, int n0) {
    const int lane  = threadIdx.x & 31;
    const int mr    = lane & 15;
    const bool vld  = mr < BB;
    const int h8    = (lane >> 4) * 8;
    const int kb16  = (lane >> 4) * 16;
    const __bf16* Ab = vld ? (sA + (size_t)mr * K) : s_zero;
    const __bf16* Wb = reinterpret_cast<const __bf16*>(WT) +
                       (size_t)(n0 + mr) * (size_t)K + kb16;
    v8bf lo = *reinterpret_cast<const v8bf*>(Ab + (vld ? kb16 : 0));
    v8bf hi = *reinterpret_cast<const v8bf*>(Ab + (vld ? kb16 + 8 : 0));
    v16bf bc = *reinterpret_cast<const v16bf*>(Wb);
    v8f c = {};
    for (int k0 = 0; k0 < K; k0 += 32) {
        int kn = (k0 + 32 < K) ? (k0 + 32) : k0;
        v8bf  lon = *reinterpret_cast<const v8bf*>(Ab + (vld ? kn + kb16 : 0));
        v8bf  hin = *reinterpret_cast<const v8bf*>(Ab + (vld ? kn + kb16 + 8 : 0));
        v16bf bn  = *reinterpret_cast<const v16bf*>(Wb + kn);
        v16bf a = __builtin_shufflevector(lo, hi, 0, 1, 2, 3, 4, 5, 6, 7,
                                          8, 9, 10, 11, 12, 13, 14, 15);
        c = __builtin_amdgcn_wmma_f32_16x16x32_bf16(false, a, false, bc,
                                                    (short)0, c, false, false);
        lo = lon; hi = hin; bc = bn;
    }
    return c;
}

// ---------------------------------------------------------------- kernels
// fp32 (K x N) -> bf16 (N x K) transpose-cast, LDS tiled.
__global__ __launch_bounds__(256) void transpose_cast(const float* __restrict__ W,
                                                      __hip_bfloat16* __restrict__ WT,
                                                      int K, int N) {
    __shared__ float tile[32][33];
    int n0 = blockIdx.x * 32, k0 = blockIdx.y * 32;
    int tx = threadIdx.x & 31, ty = threadIdx.x >> 5;
    for (int i = ty; i < 32; i += 8) {
        int k = k0 + i, n = n0 + tx;
        tile[i][tx] = (k < K && n < N) ? W[(size_t)k * N + n] : 0.0f;
    }
    __syncthreads();
    for (int i = ty; i < 32; i += 8) {
        int n = n0 + i, k = k0 + tx;
        if (n < N && k < K) WT[(size_t)n * K + k] = __float2bfloat16(tile[tx][i]);
    }
}

__global__ __launch_bounds__(256) void embed_kernel(const int* __restrict__ ids,
                                                    const float* __restrict__ emb,
                                                    float* __restrict__ x_all,
                                                    __hip_bfloat16* __restrict__ x_allb) {
    int row = blockIdx.x;                  // row = b*SS + s
    int id  = ids[row];
    for (int c = threadIdx.x; c < D; c += 256) {
        float v = emb[(size_t)id * D + c] * 32.0f;   // sqrt(1024)
        x_all[(size_t)row * D + c]  = v;
        x_allb[(size_t)row * D + c] = __float2bfloat16(v);
    }
}

// out(M x N) = A(M x K bf16) @ WT(N x K)^T (+bias). One block stages a 16-row
// A tile in LDS (via the Tensor Data Mover when available); 8 waves each
// produce a 16x64 strip; K-loop software-pipelined.
__global__ __launch_bounds__(256) void tile_gemm(const __hip_bfloat16* __restrict__ Ab16,
                                                 const __hip_bfloat16* __restrict__ WT,
                                                 const float* __restrict__ bias,
                                                 float* __restrict__ out,
                                                 int N, int ngrp) {
    __shared__ __align__(16) __bf16 sA[16 * D];   // 32KB
    const int tid  = threadIdx.x;
    const int lane = tid & 31;
    const int wv   = tid >> 5;
    int m16 = blockIdx.x / ngrp;
    int grp = blockIdx.x % ngrp;
    {   // stage A tile (16 x D bf16, contiguous 32KB)
        const __bf16* src = reinterpret_cast<const __bf16*>(Ab16) + (size_t)m16 * 16 * D;
#if HAVE_TDM
        if (tid < 32) {   // one wave issues the tensor DMA (EXEC-independent)
            unsigned lds_off = (unsigned)(uintptr_t)(void*)sA;
            unsigned long long ga = (unsigned long long)(uintptr_t)src;
            const unsigned NE = 16u * D;           // elements (2B each)
            v4u g0;
            g0[0] = 1u;                                         // count=1, user D#
            g0[1] = lds_off;                                    // LDS byte address
            g0[2] = (unsigned)(ga & 0xFFFFFFFFull);             // global_addr lo
            g0[3] = (unsigned)((ga >> 32) & 0x01FFFFFFull) | (2u << 30); // hi|type=2
            v8i g1;
            g1[0] = (int)(1u << 16);                            // data_size=2B
            g1[1] = (int)((NE & 0xFFFFu) << 16);                // tensor_dim0 lo16
            g1[2] = (int)((NE >> 16) | (1u << 16));             // dim0 hi | dim1 lo
            g1[3] = (int)((NE & 0xFFFFu) << 16);                // dim1 hi=0 | tile_dim0
            g1[4] = 1;                                          // tile_dim1=1
            g1[5] = (int)NE;                                    // dim0 stride lo32
            g1[6] = 0;
            g1[7] = 0;
            v4i g2 = {0, 0, 0, 0};
            v4i g3 = {0, 0, 0, 0};
            v8i g4 = {0, 0, 0, 0, 0, 0, 0, 0};
            __builtin_amdgcn_tensor_load_to_lds(g0, g1, g2, g3, g4, 0);
            __builtin_amdgcn_s_wait_tensorcnt(0);
        }
#else
        const v8bf* s8 = reinterpret_cast<const v8bf*>(src);
        v8bf* d8 = reinterpret_cast<v8bf*>(sA);
        for (int i = tid; i < 16 * D / 8; i += 256) d8[i] = s8[i];
#endif
    }
    __syncthreads();
    int ng = grp * 8 + wv;
    if (ng >= N / 64) return;
    int n0 = ng * 64;
    const int mr   = lane & 15;
    const int h8   = (lane >> 4) * 8;
    const int kb16 = (lane >> 4) * 16;
    const __bf16* Ar = sA + (size_t)mr * D;
    const __bf16* Bb = reinterpret_cast<const __bf16*>(WT);
    const __bf16* B0 = Bb + (size_t)(n0 + mr) * D + kb16;
    const __bf16* B1 = Bb + (size_t)(n0 + 16 + mr) * D + kb16;
    const __bf16* B2 = Bb + (size_t)(n0 + 32 + mr) * D + kb16;
    const __bf16* B3 = Bb + (size_t)(n0 + 48 + mr) * D + kb16;

    v8bf lo = *reinterpret_cast<const v8bf*>(Ar + kb16);
    v8bf hi = *reinterpret_cast<const v8bf*>(Ar + kb16 + 8);
    v16bf b0 = *reinterpret_cast<const v16bf*>(B0);
    v16bf b1 = *reinterpret_cast<const v16bf*>(B1);
    v16bf b2 = *reinterpret_cast<const v16bf*>(B2);
    v16bf b3 = *reinterpret_cast<const v16bf*>(B3);
    v8f c0 = {}, c1 = {}, c2 = {}, c3 = {};
    for (int k0 = 0; k0 < D; k0 += 32) {
        int kn = (k0 + 32 < D) ? k0 + 32 : k0;
        __builtin_prefetch(B0 + k0 + 256, 0, 1);   // global_prefetch_b8
        v8bf  lon = *reinterpret_cast<const v8bf*>(Ar + kn + kb16);
        v8bf  hin = *reinterpret_cast<const v8bf*>(Ar + kn + kb16 + 8);
        v16bf b0n = *reinterpret_cast<const v16bf*>(B0 + kn);
        v16bf b1n = *reinterpret_cast<const v16bf*>(B1 + kn);
        v16bf b2n = *reinterpret_cast<const v16bf*>(B2 + kn);
        v16bf b3n = *reinterpret_cast<const v16bf*>(B3 + kn);
        v16bf a = __builtin_shufflevector(lo, hi, 0, 1, 2, 3, 4, 5, 6, 7,
                                          8, 9, 10, 11, 12, 13, 14, 15);
        c0 = __builtin_amdgcn_wmma_f32_16x16x32_bf16(false, a, false, b0, (short)0, c0, false, false);
        c1 = __builtin_amdgcn_wmma_f32_16x16x32_bf16(false, a, false, b1, (short)0, c1, false, false);
        c2 = __builtin_amdgcn_wmma_f32_16x16x32_bf16(false, a, false, b2, (short)0, c2, false, false);
        c3 = __builtin_amdgcn_wmma_f32_16x16x32_bf16(false, a, false, b3, (short)0, c3, false, false);
        lo = lon; hi = hin; b0 = b0n; b1 = b1n; b2 = b2n; b3 = b3n;
    }
    int rbase = m16 * 16 + 8 * (lane >> 4);
#pragma unroll
    for (int j = 0; j < 8; ++j) {
        size_t r = (size_t)(rbase + j) * (size_t)N;
        float bj0 = bias ? bias[n0 + mr]      : 0.0f;
        float bj1 = bias ? bias[n0 + 16 + mr] : 0.0f;
        float bj2 = bias ? bias[n0 + 32 + mr] : 0.0f;
        float bj3 = bias ? bias[n0 + 48 + mr] : 0.0f;
        out[r + n0 + mr]      = c0[j] + bj0;
        out[r + n0 + 16 + mr] = c1[j] + bj1;
        out[r + n0 + 32 + mr] = c2[j] + bj2;
        out[r + n0 + 48 + mr] = c3[j] + bj3;
    }
}

struct ScanArgs {
    const float *lw, *kb, *vb, *qb, *ob, *f1b, *f2b, *n1w, *n1b, *n2w, *n2b;
    const __hip_bfloat16 *qWT, *oWT, *f1WT, *f2WT, *kWT, *vWT;
    const float *q0buf;              // precomputed layer-0 q for all (b,t)
    float *x_all;                    // pristine embeddings
    float *hid;                      // hidden states out
    float *memk, *memv;
    float *xcur, *resid, *outs, *abuf, *hbuf, *qbuf;
    int   *bar;
};

// Skinny-GEMM phase: out[0..3][n] = epilogue(A(4xK) @ WT(NxK)^T).
__device__ __forceinline__ void mm_phase(const float* __restrict__ X, size_t xrs, int K,
                                         const __hip_bfloat16* __restrict__ WT, int ntiles,
                                         const float* __restrict__ bias,
                                         const float* __restrict__ addX, size_t ars,
                                         float* __restrict__ out, size_t ors,
                                         bool relu,
                                         __bf16* sA, const __bf16* s_zero) {
    const int tid = threadIdx.x;
    if ((int)blockIdx.x * 8 < ntiles) {
        for (int r = 0; r < BB; ++r)
            for (int k = tid; k < K; k += 256)
                sA[r * K + k] = tobf(X[(size_t)r * xrs + k]);
    }
    __syncthreads();
    const int gw = blockIdx.x * 8 + (tid >> 5);
    if (gw < ntiles) {
        const int n0 = gw * 16;
        v8f c = wmma_tile16(sA, s_zero, WT, K, n0);
        const int lane = tid & 31;
        if (lane < 16) {
            int n = n0 + lane;
#pragma unroll
            for (int j = 0; j < BB; ++j) {
                float v = c[j] + (bias ? bias[n] : 0.0f);
                if (addX) v += addX[(size_t)j * ars + n];
                if (relu) v = v > 0.0f ? v : 0.0f;
                out[(size_t)j * ors + n] = v;
            }
        }
    }
}

__device__ void ln_phase(const ScanArgs& A, const float* w, const float* b,
                         float* outs_l, float* s_red) {
    int wg = blockIdx.x, tid = threadIdx.x;
    if (wg >= BB) return;   // non-participants hit the global barrier outside
    const float* r = A.resid + (size_t)wg * D;
    float s = 0.0f;
    for (int k = tid; k < D; k += 256) s += r[k];
    float mu = block_sum(s, s_red) * (1.0f / D);
    float vv = 0.0f;
    for (int k = tid; k < D; k += 256) { float d = r[k] - mu; vv += d * d; }
    float inv = rsqrtf(block_sum(vv, s_red) * (1.0f / D) + 1e-5f);
    for (int k = tid; k < D; k += 256) {
        float y = (r[k] - mu) * inv * w[k] + b[k];
        A.xcur[(size_t)wg * D + k] = y;
        if (outs_l) outs_l[(size_t)wg * D + k] = y;
    }
}

// End-of-step phase: WGs 0..15 stage m = sum(w_i * outs_i) inline (bf16 into
// LDS) then run both k- and v-projections (128 wave tiles); WGs 16..19 write
// the step's hidden row. One global barrier total.
__device__ __forceinline__ void kv_phase(const ScanArgs& A, int t, const float* w5,
                                         __bf16* sA, const __bf16* s_zero) {
    const int tid = threadIdx.x;
    const int wg  = blockIdx.x;
    if (wg < 16) {
        for (int r = 0; r < BB; ++r) {
            const float* x0 = A.x_all + ((size_t)r * SS + t) * D;
            for (int k = tid; k < D; k += 256) {
                float m = w5[0] * x0[k];
#pragma unroll
                for (int l = 0; l < NL; ++l)
                    m += w5[l + 1] * A.outs[((size_t)l * BB + r) * D + k];
                sA[r * D + k] = tobf(m);
            }
        }
    } else if (wg < 16 + BB) {
        int b = wg - 16;
        for (int k = tid; k < D; k += 256)
            A.hid[((size_t)b * SS + t) * D + k] = A.xcur[(size_t)b * D + k];
    }
    __syncthreads();
    const int gw = wg * 8 + (tid >> 5);
    if (gw < 128) {
        int isv = gw >= 64;
        int n0 = (gw & 63) * 16;
        const __hip_bfloat16* WT = isv ? A.vWT : A.kWT;
        const float* bias = isv ? A.vb : A.kb;
        float* out = (isv ? A.memv : A.memk) + (size_t)t * D;
        v8f c = wmma_tile16(sA, s_zero, WT, D, n0);
        const int lane = tid & 31;
        if (lane < 16) {
            int n = n0 + lane;
#pragma unroll
            for (int j = 0; j < BB; ++j)
                out[(size_t)j * SS * D + n] = c[j] + bias[n];
        }
    }
}

__global__ __launch_bounds__(256) void scan_kernel(ScanArgs A) {
    __shared__ float s_red[256];
    __shared__ float s_q[HD];
    __shared__ float s_w[SS];
    __shared__ __align__(16) __bf16 sA[BB * FF];   // staged A fragments (32KB max)
    __shared__ __align__(16) __bf16 s_zero[16];
    const int tid = threadIdx.x;
    const int wg  = blockIdx.x;
    if (tid < 16) s_zero[tid] = tobf(0.0f);
    __syncthreads();

    float w5[NL + 1];
    {
        float mx = A.lw[0];
        for (int i = 1; i <= NL; ++i) mx = fmaxf(mx, A.lw[i]);
        float sm = 0.0f;
        for (int i = 0; i <= NL; ++i) { w5[i] = __expf(A.lw[i] - mx); sm += w5[i]; }
        for (int i = 0; i <= NL; ++i) w5[i] /= sm;
    }

    for (int t = 0; t < SS; ++t) {
        const float* xin = A.x_all + (size_t)t * D;    // row b at stride SS*D
        if (t > 0) {
            for (int l = 0; l < NL; ++l) {
                const float* xs = (l == 0) ? xin : A.xcur;
                size_t xrs = (l == 0) ? (size_t)SS * D : (size_t)D;

                // ---- q projection (layer 0 precomputed outside the scan) -
                if (l > 0) {
                    mm_phase(A.xcur, D, D, A.qWT + (size_t)l * D * D, D / 16,
                             A.qb + (size_t)l * D, nullptr, 0,
                             A.qbuf, D, false, sA, s_zero);
                    gbar(A.bar);
                }

                // ---- attention: one WG per (b,h) -------------------------
                if (wg < BB * NH) {
                    int b = wg >> 4, h = wg & 15;
                    const float* qrow = (l == 0)
                        ? (A.q0buf + ((size_t)b * SS + t) * D)
                        : (A.qbuf + (size_t)b * D);
                    if (tid < HD) s_q[tid] = qrow[h * HD + tid];
                    __syncthreads();
                    float slope = exp2f(-0.5f * (float)(h + 1));
                    float lmax = -1e30f;
                    for (int s0 = tid; s0 < t; s0 += 256) {
                        const float* kr = A.memk + ((size_t)b * SS + s0) * D + h * HD;
                        float lg = 0.0f;
                        for (int c = 0; c < HD; ++c) lg += s_q[c] * kr[c];
                        lg = lg * 0.125f - slope * fabsf((float)(t - 1 - s0));
                        s_w[s0] = lg;
                        lmax = fmaxf(lmax, lg);
                    }
                    float gmax = block_max(lmax, s_red);
                    float lsum = 0.0f;
                    for (int s0 = tid; s0 < t; s0 += 256) {
                        float e = __expf(s_w[s0] - gmax);
                        s_w[s0] = e; lsum += e;
                    }
                    float rinv = 1.0f / block_sum(lsum, s_red);
                    __syncthreads();
                    int c = tid & 63, g = tid >> 6;
                    float acc = 0.0f;
                    for (int s0 = g; s0 < t; s0 += 4)
                        acc += s_w[s0] * A.memv[((size_t)b * SS + s0) * D + h * HD + c];
                    s_red[tid] = acc; __syncthreads();
                    if (tid < HD) {
                        float r = (s_red[tid] + s_red[tid + 64]) +
                                  (s_red[tid + 128] + s_red[tid + 192]);
                        A.abuf[(size_t)b * D + h * HD + tid] = r * rinv;
                    }
                }
                gbar(A.bar);

                // ---- o-proj + residual ----------------------------------
                mm_phase(A.abuf, D, D, A.oWT + (size_t)l * D * D, D / 16,
                         A.ob + (size_t)l * D, xs, xrs,
                         A.resid, D, false, sA, s_zero);
                gbar(A.bar);
                ln_phase(A, A.n1w + (size_t)l * D, A.n1b + (size_t)l * D,
                         nullptr, s_red);
                gbar(A.bar);

                // ---- FFN ------------------------------------------------
                mm_phase(A.xcur, D, D, A.f1WT + (size_t)l * FF * D, FF / 16,
                         A.f1b + (size_t)l * FF, nullptr, 0,
                         A.hbuf, FF, true, sA, s_zero);
                gbar(A.bar);
                mm_phase(A.hbuf, FF, FF, A.f2WT + (size_t)l * D * FF, D / 16,
                         A.f2b + (size_t)l * D, A.xcur, D,
                         A.resid, D, false, sA, s_zero);
                gbar(A.bar);
                ln_phase(A, A.n2w + (size_t)l * D, A.n2b + (size_t)l * D,
                         A.outs + (size_t)l * BB * D, s_red);
                gbar(A.bar);
            }
        } else {
            for (int l = 0; l < NL; ++l) {
                const float* xs = (l == 0) ? xin : A.xcur;
                size_t xrs = (l == 0) ? (size_t)SS * D : (size_t)D;
                mm_phase(xs, xrs, D, A.f1WT + (size_t)l * FF * D, FF / 16,
                         A.f1b + (size_t)l * FF, nullptr, 0,
                         A.hbuf, FF, true, sA, s_zero);
                gbar(A.bar);
                mm_phase(A.hbuf, FF, FF, A.f2WT + (size_t)l * D * FF, D / 16,
                         A.f2b + (size_t)l * D, xs, xrs,
                         A.resid, D, false, sA, s_zero);
                gbar(A.bar);
                ln_phase(A, A.n2w + (size_t)l * D, A.n2b + (size_t)l * D,
                         A.outs + (size_t)l * BB * D, s_red);
                gbar(A.bar);
            }
        }
        // ---- merged m + k/v projections + hidden store (one phase) ------
        kv_phase(A, t, w5, sA, s_zero);
        gbar(A.bar);
    }
}

__global__ __launch_bounds__(256) void final_ln(const float* __restrict__ hid,
                                                const float* __restrict__ w,
                                                const float* __restrict__ b,
                                                __hip_bfloat16* __restrict__ outb) {
    __shared__ float s_red[256];
    int row = blockIdx.x, tid = threadIdx.x;
    const float* r = hid + (size_t)row * D;
    float s = 0.0f;
    for (int k = tid; k < D; k += 256) s += r[k];
    float mu = block_sum(s, s_red) * (1.0f / D);
    float vv = 0.0f;
    for (int k = tid; k < D; k += 256) { float d = r[k] - mu; vv += d * d; }
    float inv = rsqrtf(block_sum(vv, s_red) * (1.0f / D) + 1e-5f);
    for (int k = tid; k < D; k += 256)
        outb[(size_t)row * D + k] = __float2bfloat16((r[k] - mu) * inv * w[k] + b[k]);
}

// ---------------------------------------------------------------- launch
extern "C" void kernel_launch(void* const* d_in, const int* in_sizes, int n_in,
                              void* d_out, int out_size, void* d_ws, size_t ws_size,
                              hipStream_t stream) {
    const int*   ids   = (const int*)  d_in[0];
    const float* emb   = (const float*)d_in[1];
    const float* lw    = (const float*)d_in[2];
    const float* kW    = (const float*)d_in[3];
    const float* kb    = (const float*)d_in[4];
    const float* vW    = (const float*)d_in[5];
    const float* vb    = (const float*)d_in[6];
    const float* qW    = (const float*)d_in[7];
    const float* qb    = (const float*)d_in[8];
    const float* oW    = (const float*)d_in[9];
    const float* ob    = (const float*)d_in[10];
    const float* f1W   = (const float*)d_in[11];
    const float* f1b   = (const float*)d_in[12];
    const float* f2W   = (const float*)d_in[13];
    const float* f2b   = (const float*)d_in[14];
    const float* n1w   = (const float*)d_in[15];
    const float* n1b   = (const float*)d_in[16];
    const float* n2w   = (const float*)d_in[17];
    const float* n2b   = (const float*)d_in[18];
    const float* lnf_w = (const float*)d_in[19];
    const float* lnf_b = (const float*)d_in[20];
    const float* headW = (const float*)d_in[21];

    char* ws = (char*)d_ws;
    size_t off = 0;
    auto alloc = [&](size_t bytes) {
        size_t o = off; off += (bytes + 255) & ~(size_t)255; return o;
    };
    __hip_bfloat16* qWT  = (__hip_bfloat16*)(ws + alloc((size_t)NL * D * D * 2));
    __hip_bfloat16* oWT  = (__hip_bfloat16*)(ws + alloc((size_t)NL * D * D * 2));
    __hip_bfloat16* f1WT = (__hip_bfloat16*)(ws + alloc((size_t)NL * FF * D * 2));
    __hip_bfloat16* f2WT = (__hip_bfloat16*)(ws + alloc((size_t)NL * D * FF * 2));
    __hip_bfloat16* kWT  = (__hip_bfloat16*)(ws + alloc((size_t)D * D * 2));
    __hip_bfloat16* vWT  = (__hip_bfloat16*)(ws + alloc((size_t)D * D * 2));
    __hip_bfloat16* hWT  = (__hip_bfloat16*)(ws + alloc((size_t)VO * D * 2));
    float* x_all = (float*)(ws + alloc((size_t)BB * SS * D * 4));
    float* hid   = (float*)(ws + alloc((size_t)BB * SS * D * 4));
    float* q0buf = (float*)(ws + alloc((size_t)BB * SS * D * 4));
    float* memk  = (float*)(ws + alloc((size_t)BB * SS * D * 4));
    float* memv  = (float*)(ws + alloc((size_t)BB * SS * D * 4));
    float* xcur  = (float*)(ws + alloc((size_t)BB * D * 4));
    float* resid = (float*)(ws + alloc((size_t)BB * D * 4));
    float* outs  = (float*)(ws + alloc((size_t)NL * BB * D * 4));
    float* abuf  = (float*)(ws + alloc((size_t)BB * D * 4));
    float* hbuf  = (float*)(ws + alloc((size_t)BB * FF * 4));
    float* qbuf  = (float*)(ws + alloc((size_t)BB * D * 4));
    int*   bar   = (int*)  (ws + alloc(2048));
    __hip_bfloat16* hidb  = (__hip_bfloat16*)memk;   // alias: memk dead post-scan
    __hip_bfloat16* x_allb = (__hip_bfloat16*)memv;  // alias: used pre-scan only

    dim3 blk(256);
    auto T = [&](const float* src, __hip_bfloat16* dst, int K, int N) {
        dim3 g((N + 31) / 32, (K + 31) / 32);
        transpose_cast<<<g, blk, 0, stream>>>(src, dst, K, N);
    };
    for (int l = 0; l < NL; ++l) {
        T(qW + (size_t)l * D * D,   qWT  + (size_t)l * D * D,  D,  D);
        T(oW + (size_t)l * D * D,   oWT  + (size_t)l * D * D,  D,  D);
        T(f1W + (size_t)l * D * FF, f1WT + (size_t)l * FF * D, D,  FF);
        T(f2W + (size_t)l * FF * D, f2WT + (size_t)l * D * FF, FF, D);
    }
    T(kW, kWT, D, D);
    T(vW, vWT, D, D);
    T(headW, hWT, D, VO);

    embed_kernel<<<BB * SS, blk, 0, stream>>>(ids, emb, x_all, x_allb);
    // Pre-scan layer-0 q for all timesteps: (2048 x 1024) @ qW[0].
    tile_gemm<<<((BB * SS) / 16) * 2, blk, 0, stream>>>(x_allb, qWT, qb, q0buf,
                                                        D, 2);
    (void)hipMemsetAsync(memk, 0, (size_t)BB * SS * D * 4, stream);
    (void)hipMemsetAsync(memv, 0, (size_t)BB * SS * D * 4, stream);
    (void)hipMemsetAsync(bar,  0, 2048, stream);

    ScanArgs A;
    A.lw = lw; A.kb = kb; A.vb = vb; A.qb = qb; A.ob = ob;
    A.f1b = f1b; A.f2b = f2b; A.n1w = n1w; A.n1b = n1b; A.n2w = n2w; A.n2b = n2b;
    A.qWT = qWT; A.oWT = oWT; A.f1WT = f1WT; A.f2WT = f2WT; A.kWT = kWT; A.vWT = vWT;
    A.q0buf = q0buf; A.x_all = x_all; A.hid = hid; A.memk = memk; A.memv = memv;
    A.xcur = xcur; A.resid = resid; A.outs = outs;
    A.abuf = abuf; A.hbuf = hbuf; A.qbuf = qbuf; A.bar = bar;
    scan_kernel<<<NWG, blk, 0, stream>>>(A);

    final_ln<<<BB * SS, blk, 0, stream>>>(hid, lnf_w, lnf_b, hidb);
    tile_gemm<<<((BB * SS) / 16) * ((VO / 64 + 7) / 8), blk, 0, stream>>>(
        hidb, hWT, nullptr, (float*)d_out, VO, (VO / 64 + 7) / 8);
}
